// TransformerLayer_42898133353385
// MI455X (gfx1250) — matrix operand order repaired and checked by reading, not compiled, not run on hardware
//
#include <hip/hip_runtime.h>
#include <hip/hip_bf16.h>
#include <math.h>

#define S_   2048
#define D_   1024
#define H_   16
#define DI_  64
#define DFF_ 4096

typedef __bf16 bf16_t;
typedef __attribute__((ext_vector_type(16))) __bf16 v16bf;
typedef __attribute__((ext_vector_type(8)))  __bf16 bf16x8;
typedef __attribute__((ext_vector_type(8)))  float  v8f;

// ---------------------------------------------------------------------------
// WMMA fragment loaders (wave32, 16-bit operands); k order below is this code's, same for A and B
// A (16x32, MxK): lane L -> m = L&15, half g = L>>4.
//   lane holds K = g*16 .. g*16+15 contiguous (packed pairs).
// B (32x16, KxN): lane L -> n = L&15, half g = L>>4.
//   lane holds K = g*16 .. g*16+15 contiguous (packed pairs).
// C/D (16x16 f32): lane L -> n = L&15, VGPR r -> m = (L<16 ? r : r+8).
// ---------------------------------------------------------------------------
static __device__ inline v16bf frag_a(const bf16_t* A, int lda, int m0, int k0, int lane) {
  int m = lane & 15, g = lane >> 4;
  const bf16_t* row = A + (size_t)(m0 + m) * lda + k0 + g * 16;
  bf16x8 lo = *(const bf16x8*)(row);
  bf16x8 hi = *(const bf16x8*)(row + 8);
  v16bf a;
#pragma unroll
  for (int i = 0; i < 8; ++i) { a[i] = lo[i]; a[8 + i] = hi[i]; }
  return a;
}

static __device__ inline v16bf frag_b(const bf16_t* Bt, int ldb, int n0, int k0, int lane) {
  int n = lane & 15, g = lane >> 4;
  return *(const v16bf*)(Bt + (size_t)(n0 + n) * ldb + k0 + g * 16);
}

// B fragment from a staged LDS chunk laid out [64][32] (n-major, k contiguous).
static __device__ inline v16bf frag_b_lds(const bf16_t* chunk, int lane) {
  int n = lane & 15, g = lane >> 4;
  return *(const v16bf*)(chunk + n * 32 + g * 16);
}

static __device__ inline v8f wmma_bf16(v16bf a, v16bf b, v8f c) {
  return __builtin_amdgcn_wmma_f32_16x16x32_bf16(false, a, false, b, (short)0, c, false, false);
}

// Async-copy one 16B segment of the 64x32 B chunk into LDS (per thread).
// Generic LDS pointers carry the LDS byte offset in addr[31:0] (ISA 10.2),
// which is exactly the VDST operand global_load_async_to_lds expects.
static __device__ inline void async_b_chunk(const bf16_t* Bt, int ldb, int n0, int kc,
                                            bf16_t (*bstage)[64][32], int buf,
                                            int row, int seg) {
  const bf16_t* src = Bt + (size_t)(n0 + row) * ldb + kc * 32 + seg * 16;
  unsigned lds = (unsigned)(uintptr_t)(const void*)&bstage[buf][row][seg * 16];
  unsigned long long g = (unsigned long long)(uintptr_t)src;
  asm volatile("global_load_async_to_lds_b128 %0, %1, off"
               :: "v"(lds), "v"(g) : "memory");
}

// ---------------------------------------------------------------------------
// Block-level GEMM: 8 waves, wave w owns rows m0+w*16..+15; all waves share the
// 64-column B panel [n0..n0+63], double-buffered through LDS via ASYNC copies.
// C = A(row-major) * Bt^T  (Bt is [N][K] row-major).
// ---------------------------------------------------------------------------
static __device__ inline void block_gemm4(const bf16_t* __restrict__ A, int lda,
                                          const bf16_t* __restrict__ Bt, int ldb,
                                          int m0, int n0, int K, int tid,
                                          bf16_t (*bstage)[64][32], v8f acc[4]) {
  int lane = tid & 31;
  int row = tid >> 2, seg = tid & 3;          // 256 threads cover 64 rows x 4 segs
  int nk = K >> 5;
  async_b_chunk(Bt, ldb, n0, 0, bstage, 0, row, seg);
  for (int kc = 0; kc < nk; ++kc) {
    int buf = kc & 1;
    asm volatile("s_wait_asynccnt 0x0" ::: "memory");
    __syncthreads();                           // buf ready; buf^1 fully consumed
    if (kc + 1 < nk)
      async_b_chunk(Bt, ldb, n0, kc + 1, bstage, buf ^ 1, row, seg);
    v16bf a = frag_a(A, lda, m0, kc * 32, lane);
    __builtin_prefetch(A + (size_t)(m0 + (lane & 15)) * lda + kc * 32 + 64, 0, 1);
#pragma unroll
    for (int t = 0; t < 4; ++t) {
      v16bf b = frag_b_lds(&bstage[buf][t * 16][0], lane);
      acc[t] = wmma_bf16(a, b, acc[t]);
    }
  }
}

// ---------------------------------------------------------------------------
// Conversions
// ---------------------------------------------------------------------------
__global__ void k_cvt_bf16(const float* __restrict__ src, bf16_t* __restrict__ dst, int n) {
  for (int i = blockIdx.x * blockDim.x + threadIdx.x; i < n; i += gridDim.x * blockDim.x)
    dst[i] = (bf16_t)src[i];
}

// src [H][D][64] fp32 -> dst [H][64][D] bf16   (n = H*64*D)
__global__ void k_cvt_tr(const float* __restrict__ src, bf16_t* __restrict__ dst, int n) {
  for (int i = blockIdx.x * blockDim.x + threadIdx.x; i < n; i += gridDim.x * blockDim.x) {
    int h = i >> 16;            // 64*1024
    int r = i & 65535;
    int e = r >> 10;
    int d = r & 1023;
    dst[i] = (bf16_t)src[((size_t)h * D_ + d) * DI_ + e];
  }
}

// ---------------------------------------------------------------------------
// QKV projections: grid (S/128, H, 3).  Q,K row-major [h][S][64]; V transposed.
// ---------------------------------------------------------------------------
__global__ __launch_bounds__(256) void k_qkv(const bf16_t* __restrict__ Xb,
                                             const bf16_t* __restrict__ WQt,
                                             const bf16_t* __restrict__ WKt,
                                             const bf16_t* __restrict__ WVt,
                                             bf16_t* __restrict__ Qb,
                                             bf16_t* __restrict__ Kb,
                                             bf16_t* __restrict__ Vtb) {
  __shared__ bf16_t bstage[2][64][32];
  int tid = threadIdx.x, lane = tid & 31, wave = tid >> 5;
  int r0 = blockIdx.x * 128 + wave * 16;
  int h = blockIdx.y, kind = blockIdx.z;
  const bf16_t* Wt = (kind == 0 ? WQt : kind == 1 ? WKt : WVt) + (size_t)h * DI_ * D_;
  v8f acc[4] = {};
  block_gemm4(Xb, D_, Wt, D_, r0, 0, D_, tid, bstage, acc);
  if (kind < 2) {
    bf16_t* out = (kind == 0 ? Qb : Kb) + (size_t)h * S_ * DI_;
#pragma unroll
    for (int t = 0; t < 4; ++t)
#pragma unroll
      for (int r = 0; r < 8; ++r) {
        int m = (lane < 16) ? r : r + 8;
        int n = t * 16 + (lane & 15);
        out[(size_t)(r0 + m) * DI_ + n] = (bf16_t)acc[t][r];
      }
  } else {
    bf16_t* out = Vtb + (size_t)h * DI_ * S_;
#pragma unroll
    for (int t = 0; t < 4; ++t)
#pragma unroll
      for (int r = 0; r < 8; ++r) {
        int m = (lane < 16) ? r : r + 8;
        int n = t * 16 + (lane & 15);
        out[(size_t)n * S_ + (r0 + m)] = (bf16_t)acc[t][r];
      }
  }
}

// ---------------------------------------------------------------------------
// Fused attention: grid (S/16, H), 256 threads (8 waves).
// scores(WMMA) -> +ALiBi +causal -> softmax -> SA(fp32, d_out) -> z = P @ V
// ---------------------------------------------------------------------------
__global__ __launch_bounds__(256) void k_attn(const bf16_t* __restrict__ Qb,
                                              const bf16_t* __restrict__ Kb,
                                              const bf16_t* __restrict__ Vtb,
                                              float* __restrict__ SA,
                                              bf16_t* __restrict__ Zb) {
  __shared__ bf16_t sc[16][S_];       // 64 KB: score/prob tile
  __shared__ float  zp[8][16][64];    // 32 KB: per-wave z partials
  __shared__ float  red[16][16];
  __shared__ float  rowstat[16];

  int tid = threadIdx.x, lane = tid & 31, wave = tid >> 5;
  int r0 = blockIdx.x * 16, h = blockIdx.y;
  const bf16_t* Qh = Qb + (size_t)h * S_ * DI_;
  const bf16_t* Kh = Kb + (size_t)h * S_ * DI_;
  const bf16_t* Vh = Vtb + (size_t)h * DI_ * S_;
  float slope = exp2f(-0.5f * (float)(h + 1));   // (2^-8/H)^(h+1)

  // ---- phase 1: scores = (Q K^T)/8 + bias + mask, stored bf16 in LDS ----
  v16bf aq0 = frag_a(Qh, DI_, r0, 0, lane);
  v16bf aq1 = frag_a(Qh, DI_, r0, 32, lane);
  for (int ct = wave * 16; ct < wave * 16 + 16; ++ct) {
    v8f acc = {};
    acc = wmma_bf16(aq0, frag_b(Kh, DI_, ct * 16, 0, lane), acc);
    acc = wmma_bf16(aq1, frag_b(Kh, DI_, ct * 16, 32, lane), acc);
#pragma unroll
    for (int r = 0; r < 8; ++r) {
      int m = (lane < 16) ? r : r + 8;
      int gr = r0 + m;
      int gc = ct * 16 + (lane & 15);
      int dlt = gr - gc;
      float v = acc[r] * 0.125f - slope * (float)dlt;
      if (dlt < 0) v = -1e30f;                   // causal mask
      sc[m][gc] = (bf16_t)v;
    }
  }
  __syncthreads();

  // ---- phase 2: softmax (16 threads per row), write SA fp32 ----
  int row = tid >> 4, sub = tid & 15;
  float mx = -1e30f;
  for (int j = sub; j < S_; j += 16) mx = fmaxf(mx, (float)sc[row][j]);
  red[row][sub] = mx;
  __syncthreads();
  if (sub == 0) {
    float m2 = -1e30f;
    for (int t = 0; t < 16; ++t) m2 = fmaxf(m2, red[row][t]);
    rowstat[row] = m2;
  }
  __syncthreads();
  float rm = rowstat[row];
  float sm = 0.f;
  for (int j = sub; j < S_; j += 16) sm += __expf((float)sc[row][j] - rm);
  red[row][sub] = sm;
  __syncthreads();
  if (sub == 0) {
    float s2 = 0.f;
    for (int t = 0; t < 16; ++t) s2 += red[row][t];
    rowstat[row] = s2;
  }
  __syncthreads();
  float inv = 1.0f / rowstat[row];
  {
    float* out = SA + ((size_t)h * S_ + (r0 + row)) * S_;
    for (int j = sub; j < S_; j += 16) {
      float p = __expf((float)sc[row][j] - rm) * inv;
      out[j] = p;
      sc[row][j] = (bf16_t)p;                    // normalized P for z-GEMM
    }
  }
  __syncthreads();

  // ---- phase 3: z = P @ V, K split over 8 waves, deterministic reduce ----
  {
    v8f zacc[4] = {};
    int kw = wave * 256;
    for (int ks = 0; ks < 8; ++ks) {
      int k0 = kw + ks * 32;
      v16bf a = frag_a(&sc[0][0], S_, 0, k0, lane);   // A from LDS (ds_load)
#pragma unroll
      for (int t = 0; t < 4; ++t) {
        v16bf b = frag_b(Vh, S_, t * 16, k0, lane);
        zacc[t] = wmma_bf16(a, b, zacc[t]);
      }
    }
#pragma unroll
    for (int t = 0; t < 4; ++t)
#pragma unroll
      for (int r = 0; r < 8; ++r) {
        int m = (lane < 16) ? r : r + 8;
        int n = t * 16 + (lane & 15);
        zp[wave][m][n] = zacc[t][r];
      }
  }
  __syncthreads();
  for (int i = tid; i < 16 * 64; i += 256) {
    int rr = i >> 6, c = i & 63;
    float s = 0.f;
#pragma unroll
    for (int w = 0; w < 8; ++w) s += zp[w][rr][c];
    Zb[(size_t)(r0 + rr) * (H_ * DI_) + h * DI_ + c] = (bf16_t)s;
  }
}

// ---------------------------------------------------------------------------
// WO projection + residual: attouts = X + Z @ WO^T   (grid 16x16)
// ---------------------------------------------------------------------------
__global__ __launch_bounds__(256) void k_wo(const bf16_t* __restrict__ Zb,
                                            const bf16_t* __restrict__ WOtb,
                                            const float* __restrict__ X,
                                            const float* __restrict__ WOb,
                                            float* __restrict__ ATT,
                                            bf16_t* __restrict__ ATTb) {
  __shared__ bf16_t bstage[2][64][32];
  int tid = threadIdx.x, lane = tid & 31, wave = tid >> 5;
  int r0 = blockIdx.x * 128 + wave * 16;
  int n0 = blockIdx.y * 64;
  v8f acc[4] = {};
  block_gemm4(Zb, D_, WOtb, D_, r0, n0, D_, tid, bstage, acc);
#pragma unroll
  for (int t = 0; t < 4; ++t)
#pragma unroll
    for (int r = 0; r < 8; ++r) {
      int m = (lane < 16) ? r : r + 8;
      int gr = r0 + m, gc = n0 + t * 16 + (lane & 15);
      float c = acc[t][r] + WOb[gc] + X[(size_t)gr * D_ + gc];
      ATT[(size_t)gr * D_ + gc] = c;
      ATTb[(size_t)gr * D_ + gc] = (bf16_t)c;
    }
}

// FF1: H1 = relu(att @ FF1^T + b1), bf16 out  (grid 16x64)
__global__ __launch_bounds__(256) void k_ff1(const bf16_t* __restrict__ ATTb,
                                             const bf16_t* __restrict__ F1t,
                                             const float* __restrict__ b1,
                                             bf16_t* __restrict__ H1b) {
  __shared__ bf16_t bstage[2][64][32];
  int tid = threadIdx.x, lane = tid & 31, wave = tid >> 5;
  int r0 = blockIdx.x * 128 + wave * 16;
  int n0 = blockIdx.y * 64;
  v8f acc[4] = {};
  block_gemm4(ATTb, D_, F1t, D_, r0, n0, D_, tid, bstage, acc);
#pragma unroll
  for (int t = 0; t < 4; ++t)
#pragma unroll
    for (int r = 0; r < 8; ++r) {
      int m = (lane < 16) ? r : r + 8;
      int gr = r0 + m, gc = n0 + t * 16 + (lane & 15);
      float c = fmaxf(acc[t][r] + b1[gc], 0.f);
      H1b[(size_t)gr * DFF_ + gc] = (bf16_t)c;
    }
}

// FF2: out = att + H1 @ FF2^T + b2, fp32 out  (grid 16x16)
__global__ __launch_bounds__(256) void k_ff2(const bf16_t* __restrict__ H1b,
                                             const bf16_t* __restrict__ F2t,
                                             const float* __restrict__ b2,
                                             const float* __restrict__ ATT,
                                             float* __restrict__ OUT) {
  __shared__ bf16_t bstage[2][64][32];
  int tid = threadIdx.x, lane = tid & 31, wave = tid >> 5;
  int r0 = blockIdx.x * 128 + wave * 16;
  int n0 = blockIdx.y * 64;
  v8f acc[4] = {};
  block_gemm4(H1b, DFF_, F2t, DFF_, r0, n0, DFF_, tid, bstage, acc);
#pragma unroll
  for (int t = 0; t < 4; ++t)
#pragma unroll
    for (int r = 0; r < 8; ++r) {
      int m = (lane < 16) ? r : r + 8;
      int gr = r0 + m, gc = n0 + t * 16 + (lane & 15);
      OUT[(size_t)gr * D_ + gc] = acc[t][r] + b2[gc] + ATT[(size_t)gr * D_ + gc];
    }
}

// ---------------------------------------------------------------------------
extern "C" void kernel_launch(void* const* d_in, const int* in_sizes, int n_in,
                              void* d_out, int out_size, void* d_ws, size_t ws_size,
                              hipStream_t stream) {
  const float* X   = (const float*)d_in[0];
  const float* WQs = (const float*)d_in[1];
  const float* WKs = (const float*)d_in[2];
  const float* WVs = (const float*)d_in[3];
  const float* WOw = (const float*)d_in[4];
  const float* WOb = (const float*)d_in[5];
  const float* F1w = (const float*)d_in[6];
  const float* F1b = (const float*)d_in[7];
  const float* F2w = (const float*)d_in[8];
  const float* F2b = (const float*)d_in[9];

  float* SA   = (float*)d_out;
  float* OUTV = SA + (size_t)H_ * S_ * S_;

  char* w = (char*)d_ws;
  auto alloc = [&](size_t bytes) {
    char* p = w;
    w += (bytes + 255) & ~(size_t)255;
    return (void*)p;
  };
  bf16_t* Xb   = (bf16_t*)alloc((size_t)S_ * D_ * 2);
  bf16_t* WQt  = (bf16_t*)alloc((size_t)H_ * DI_ * D_ * 2);
  bf16_t* WKt  = (bf16_t*)alloc((size_t)H_ * DI_ * D_ * 2);
  bf16_t* WVt  = (bf16_t*)alloc((size_t)H_ * DI_ * D_ * 2);
  bf16_t* WOtb = (bf16_t*)alloc((size_t)D_ * D_ * 2);
  bf16_t* F1t  = (bf16_t*)alloc((size_t)DFF_ * D_ * 2);
  bf16_t* F2t  = (bf16_t*)alloc((size_t)D_ * DFF_ * 2);
  bf16_t* Qb   = (bf16_t*)alloc((size_t)H_ * S_ * DI_ * 2);
  bf16_t* Kb   = (bf16_t*)alloc((size_t)H_ * S_ * DI_ * 2);
  bf16_t* Vtb  = (bf16_t*)alloc((size_t)H_ * DI_ * S_ * 2);
  bf16_t* Zb   = (bf16_t*)alloc((size_t)S_ * D_ * 2);
  float*  ATT  = (float*)alloc((size_t)S_ * D_ * 4);
  bf16_t* ATTb = (bf16_t*)alloc((size_t)S_ * D_ * 2);
  bf16_t* H1b  = (bf16_t*)alloc((size_t)S_ * DFF_ * 2);
  (void)ws_size; (void)in_sizes; (void)n_in; (void)out_size;

  k_cvt_bf16<<<2048, 256, 0, stream>>>(X,   Xb,   S_ * D_);
  k_cvt_bf16<<<2048, 256, 0, stream>>>(WOw, WOtb, D_ * D_);      // [out,in] == Bt layout
  k_cvt_bf16<<<2048, 256, 0, stream>>>(F1w, F1t,  DFF_ * D_);
  k_cvt_bf16<<<2048, 256, 0, stream>>>(F2w, F2t,  D_ * DFF_);
  k_cvt_tr <<<2048, 256, 0, stream>>>(WQs, WQt,  H_ * DI_ * D_);
  k_cvt_tr <<<2048, 256, 0, stream>>>(WKs, WKt,  H_ * DI_ * D_);
  k_cvt_tr <<<2048, 256, 0, stream>>>(WVs, WVt,  H_ * DI_ * D_);

  k_qkv <<<dim3(S_ / 128, H_, 3), 256, 0, stream>>>(Xb, WQt, WKt, WVt, Qb, Kb, Vtb);
  k_attn<<<dim3(S_ / 16, H_),     256, 0, stream>>>(Qb, Kb, Vtb, SA, Zb);
  k_wo  <<<dim3(S_ / 128, D_ / 64),   256, 0, stream>>>(Zb, WOtb, X, WOb, ATT, ATTb);
  k_ff1 <<<dim3(S_ / 128, DFF_ / 64), 256, 0, stream>>>(ATTb, F1t, F1b, H1b);
  k_ff2 <<<dim3(S_ / 128, D_ / 64),   256, 0, stream>>>(H1b, F2t, F2b, ATT, OUTV);
}
